// MoELayer_11579231830573
// MI455X (gfx1250) — compile-verified, matrix operand order checked by reading, not run on hardware
//
#include <hip/hip_runtime.h>

// ---------------------------------------------------------------------------
// MoE (top-2 of 8 experts), T=4096 tokens, H=1024, I=4096.
// bf16 WMMA (v_wmma_f32_16x16x32_bf16) with on-the-fly fp32->bf16 conversion.
// Each wave owns two 16x16 N-tiles so every A-fragment feeds 2-4 WMMAs.
// ---------------------------------------------------------------------------

#define TOK   4096            // B*S
#define HDIM  1024
#define NEXP  8
#define IDIM  4096
#define TOPK  2
#define TOTA  (TOK * TOPK)    // total assignments = 8192

typedef __attribute__((ext_vector_type(16))) __bf16 v16bf;
typedef __attribute__((ext_vector_type(8)))  __bf16 v8bf;
typedef __attribute__((ext_vector_type(8)))  float  v8f;

static __device__ __forceinline__ v16bf wmma_pack_bf16(const __bf16* p, int kb) {
  // 16 bf16 operand halves: K = kb+0..7 and K = 16+kb+0..7  (ISA 16-bit A/B layout)
  v8bf lo = *(const v8bf*)(p + kb);
  v8bf hi = *(const v8bf*)(p + 16 + kb);
  v16bf r;
#pragma unroll
  for (int j = 0; j < 8; ++j) { r[j] = lo[j]; r[8 + j] = hi[j]; }
  return r;
}

static __device__ __forceinline__ v16bf wmma_pack_f32(const float* p, int kb) {
  // load 16 fp32 (two 8-float runs at +kb and +16+kb), convert to bf16
  float4 f0 = *(const float4*)(p + kb);
  float4 f1 = *(const float4*)(p + kb + 4);
  float4 f2 = *(const float4*)(p + kb + 16);
  float4 f3 = *(const float4*)(p + kb + 20);
  v16bf r;
  r[0]=(__bf16)f0.x;  r[1]=(__bf16)f0.y;  r[2]=(__bf16)f0.z;  r[3]=(__bf16)f0.w;
  r[4]=(__bf16)f1.x;  r[5]=(__bf16)f1.y;  r[6]=(__bf16)f1.z;  r[7]=(__bf16)f1.w;
  r[8]=(__bf16)f2.x;  r[9]=(__bf16)f2.y;  r[10]=(__bf16)f2.z; r[11]=(__bf16)f2.w;
  r[12]=(__bf16)f3.x; r[13]=(__bf16)f3.y; r[14]=(__bf16)f3.z; r[15]=(__bf16)f3.w;
  return r;
}

static __device__ __forceinline__ void atomic_add_f32(float* p, float v) {
  __hip_atomic_fetch_add(p, v, __ATOMIC_RELAXED, __HIP_MEMORY_SCOPE_AGENT);
}

static __device__ __forceinline__ float silu(float g) {
  // g * sigmoid(g) without IEEE divide: v_exp_f32 + v_rcp_f32
  return g * __builtin_amdgcn_rcpf(1.f + __expf(-g));
}

// ------------------------------- setup kernels ------------------------------

__global__ __launch_bounds__(32) void k_zero_meta(int* counts16) {
  if (threadIdx.x < 16) counts16[threadIdx.x] = 0;   // counts[8] + cursor[8]
}

__global__ __launch_bounds__(256) void k_zero_out(float4* out4) {
  out4[(size_t)blockIdx.x * 256 + threadIdx.x] = make_float4(0.f, 0.f, 0.f, 0.f);
}

__global__ __launch_bounds__(256) void k_router(const float* __restrict__ x,
                                                const float* __restrict__ gate_w,
                                                int* __restrict__ counts,
                                                int* __restrict__ topk_idx,
                                                float* __restrict__ topk_w) {
  __shared__ float gw[NEXP * HDIM];                  // 32 KB
  int tid = threadIdx.x;
  for (int j = tid; j < NEXP * HDIM; j += 256) gw[j] = gate_w[j];
  __syncthreads();
  int t = blockIdx.x * 256 + tid;
  const float* xr = x + (size_t)t * HDIM;
  float acc[NEXP];
#pragma unroll
  for (int e = 0; e < NEXP; ++e) acc[e] = 0.f;
  for (int h = 0; h < HDIM; ++h) {
    float xv = xr[h];
#pragma unroll
    for (int e = 0; e < NEXP; ++e) acc[e] = fmaf(xv, gw[e * HDIM + h], acc[e]);
  }
  int i0 = 0; float v0 = acc[0];
#pragma unroll
  for (int e = 1; e < NEXP; ++e) if (acc[e] > v0) { v0 = acc[e]; i0 = e; }
  int i1 = 0; float v1 = -3.4e38f;
#pragma unroll
  for (int e = 0; e < NEXP; ++e) if (e != i0 && acc[e] > v1) { v1 = acc[e]; i1 = e; }
  float p0 = 1.f / (1.f + __expf(v1 - v0));          // softmax over top-2
  topk_idx[2 * t] = i0;  topk_idx[2 * t + 1] = i1;
  topk_w[2 * t] = p0;    topk_w[2 * t + 1] = 1.f - p0;
  atomicAdd(&counts[i0], 1);
  atomicAdd(&counts[i1], 1);
}

__global__ void k_prefix(const int* __restrict__ counts, int* __restrict__ offsets,
                         int* __restrict__ cursor) {
  int s = 0;
  for (int e = 0; e < NEXP; ++e) { offsets[e] = s; cursor[e] = s; s += counts[e]; }
}

__global__ __launch_bounds__(256) void k_scatter(const int* __restrict__ topk_idx,
                                                 const float* __restrict__ topk_w,
                                                 int* __restrict__ cursor,
                                                 int* __restrict__ list_token,
                                                 float* __restrict__ list_w) {
  int t = blockIdx.x * 256 + threadIdx.x;
#pragma unroll
  for (int k = 0; k < TOPK; ++k) {
    int e = topk_idx[2 * t + k];
    int pos = atomicAdd(&cursor[e], 1);
    list_token[pos] = t;
    list_w[pos] = topk_w[2 * t + k];
  }
}

// ---------------------- expert gate/up (SwiGLU hidden) ----------------------
// grid: (NEXP*256, IDIM/256); block 256 = 8 waves; per block: 16 tokens x 256 I
// Each wave: two icol tiles, gate+up -> 4 WMMAs per shared A-fragment.

__global__ __launch_bounds__(256) void k_gateup(const float* __restrict__ x,
                                                const float* __restrict__ wg,
                                                const float* __restrict__ wu,
                                                const int* __restrict__ offsets,
                                                const int* __restrict__ counts,
                                                const int* __restrict__ list_token,
                                                __bf16* __restrict__ hidden) {
  constexpr int XS = HDIM + 8;                       // padded LDS row stride (halves)
  __shared__ __align__(16) __bf16 xs[16 * XS];       // ~33 KB x-tile, bf16
  __shared__ int toks[16];

  int e = blockIdx.x >> 8;
  int tile = blockIdx.x & 255;
  int off = offsets[e], cnt = counts[e];
  int row0 = tile * 16;
  if (row0 >= cnt) return;

  int tid = threadIdx.x;
  if (tid < 16) {
    int a = off + row0 + tid;
    if (a > TOTA - 1) a = TOTA - 1;
    toks[tid] = list_token[a];
  }
  __syncthreads();
  for (int idx = tid; idx < 16 * HDIM; idx += 256) {
    int r = idx >> 10, c = idx & (HDIM - 1);
    xs[r * XS + c] = (__bf16)x[(size_t)toks[r] * HDIM + c];
  }
  __syncthreads();

  int lane = tid & 31, w = tid >> 5;
  int m = lane & 15, hi = lane >> 4, kb = hi * 8;
  int icol0 = blockIdx.y * 256 + w * 32 + m;         // lane&15 = B/C column (N)
  int icol1 = icol0 + 16;
  const float* wgr0 = wg + ((size_t)e * IDIM + icol0) * HDIM;
  const float* wgr1 = wg + ((size_t)e * IDIM + icol1) * HDIM;
  const float* wur0 = wu + ((size_t)e * IDIM + icol0) * HDIM;
  const float* wur1 = wu + ((size_t)e * IDIM + icol1) * HDIM;
  const __bf16* ar = xs + m * XS;                    // lane&15 = A row (M)

  v8f accg0 = {}; v8f accg1 = {}; v8f accu0 = {}; v8f accu1 = {};
  for (int h0 = 0; h0 < HDIM; h0 += 32) {
    if (h0 + 128 < HDIM) {
      __builtin_prefetch(wgr0 + h0 + 128, 0, 3);     // global_prefetch_b8
      __builtin_prefetch(wur0 + h0 + 128, 0, 3);
      __builtin_prefetch(wgr1 + h0 + 128, 0, 3);
      __builtin_prefetch(wur1 + h0 + 128, 0, 3);
    }
    v16bf a   = wmma_pack_bf16(ar + h0, kb);
    v16bf bg0 = wmma_pack_f32(wgr0 + h0, kb);
    v16bf bu0 = wmma_pack_f32(wur0 + h0, kb);
    v16bf bg1 = wmma_pack_f32(wgr1 + h0, kb);
    v16bf bu1 = wmma_pack_f32(wur1 + h0, kb);
    accg0 = __builtin_amdgcn_wmma_f32_16x16x32_bf16(false, a, false, bg0, (short)0, accg0, false, false);
    accu0 = __builtin_amdgcn_wmma_f32_16x16x32_bf16(false, a, false, bu0, (short)0, accu0, false, false);
    accg1 = __builtin_amdgcn_wmma_f32_16x16x32_bf16(false, a, false, bg1, (short)0, accg1, false, false);
    accu1 = __builtin_amdgcn_wmma_f32_16x16x32_bf16(false, a, false, bu1, (short)0, accu1, false, false);
  }
#pragma unroll
  for (int v = 0; v < 8; ++v) {                      // C: VGPR v -> row v + 8*hi
    int r = v + 8 * hi;
    if (row0 + r < cnt) {
      __bf16* hrow = hidden + (size_t)(off + row0 + r) * IDIM;
      hrow[icol0] = (__bf16)(silu(accg0[v]) * accu0[v]);
      hrow[icol1] = (__bf16)(silu(accg1[v]) * accu1[v]);
    }
  }
}

// ------------------------------ expert down ---------------------------------
// grid: (NEXP*256, HDIM/256); per block: 16 rows x 256 H cols; scatter-add out
// Each wave: two hcol tiles -> 2 WMMAs per shared A-fragment.

__global__ __launch_bounds__(256) void k_down(const __bf16* __restrict__ hidden,
                                              const float* __restrict__ wd,
                                              const int* __restrict__ offsets,
                                              const int* __restrict__ counts,
                                              const int* __restrict__ list_token,
                                              const float* __restrict__ list_w,
                                              float* __restrict__ out) {
  __shared__ int toks[16];
  __shared__ float wts[16];
  int e = blockIdx.x >> 8;
  int tile = blockIdx.x & 255;
  int off = offsets[e], cnt = counts[e];
  int row0 = tile * 16;
  if (row0 >= cnt) return;

  int tid = threadIdx.x;
  if (tid < 16) {
    int a = off + row0 + tid;
    if (a > TOTA - 1) a = TOTA - 1;
    toks[tid] = list_token[a];
    wts[tid]  = list_w[a];
  }
  __syncthreads();

  int lane = tid & 31, w = tid >> 5;
  int m = lane & 15, hi = lane >> 4, kb = hi * 8;
  int hcol0 = blockIdx.y * 256 + w * 32 + m;
  int hcol1 = hcol0 + 16;
  int arow = off + row0 + m;
  if (arow > TOTA - 1) arow = TOTA - 1;
  const __bf16* ar = hidden + (size_t)arow * IDIM;
  const float* wdr0 = wd + ((size_t)e * HDIM + hcol0) * IDIM;
  const float* wdr1 = wd + ((size_t)e * HDIM + hcol1) * IDIM;

  v8f acc0 = {}; v8f acc1 = {};
  for (int i0 = 0; i0 < IDIM; i0 += 32) {
    if (i0 + 128 < IDIM) {
      __builtin_prefetch(wdr0 + i0 + 128, 0, 3);
      __builtin_prefetch(wdr1 + i0 + 128, 0, 3);
    }
    v16bf a  = wmma_pack_bf16(ar + i0, kb);
    v16bf b0 = wmma_pack_f32(wdr0 + i0, kb);
    v16bf b1 = wmma_pack_f32(wdr1 + i0, kb);
    acc0 = __builtin_amdgcn_wmma_f32_16x16x32_bf16(false, a, false, b0, (short)0, acc0, false, false);
    acc1 = __builtin_amdgcn_wmma_f32_16x16x32_bf16(false, a, false, b1, (short)0, acc1, false, false);
  }
#pragma unroll
  for (int v = 0; v < 8; ++v) {
    int r = v + 8 * hi;
    if (row0 + r < cnt) {
      float* orow = out + (size_t)toks[r] * HDIM;
      atomic_add_f32(orow + hcol0, acc0[v] * wts[r]);
      atomic_add_f32(orow + hcol1, acc1[v] * wts[r]);
    }
  }
}

// ------------------------------- host launch --------------------------------

extern "C" void kernel_launch(void* const* d_in, const int* in_sizes, int n_in,
                              void* d_out, int out_size, void* d_ws, size_t ws_size,
                              hipStream_t stream) {
  const float* x      = (const float*)d_in[0];
  const float* gate_w = (const float*)d_in[1];
  const float* wg     = (const float*)d_in[2];
  const float* wu     = (const float*)d_in[3];
  const float* wd     = (const float*)d_in[4];
  float* out = (float*)d_out;
  char* ws = (char*)d_ws;

  // workspace layout
  int*    counts     = (int*)(ws + 0);        // 8 ints
  int*    cursor     = (int*)(ws + 32);       // 8 ints
  int*    offsets    = (int*)(ws + 64);       // 8 ints
  int*    topk_idx   = (int*)(ws + 128);                     // 8192 ints
  float*  topk_w     = (float*)(ws + 128 + 32768);           // 8192 floats
  int*    list_token = (int*)(ws + 128 + 65536);             // 8192 ints
  float*  list_w     = (float*)(ws + 128 + 98304);           // 8192 floats
  __bf16* hidden     = (__bf16*)(ws + 131328);               // 8192*4096 bf16 = 64 MB

  k_zero_meta<<<1, 32, 0, stream>>>(counts);
  k_zero_out<<<(TOK * HDIM) / (256 * 4), 256, 0, stream>>>((float4*)out);
  k_router<<<TOK / 256, 256, 0, stream>>>(x, gate_w, counts, topk_idx, topk_w);
  k_prefix<<<1, 1, 0, stream>>>(counts, offsets, cursor);
  k_scatter<<<TOK / 256, 256, 0, stream>>>(topk_idx, topk_w, cursor, list_token, list_w);

  dim3 g4(NEXP * 256, IDIM / 256);
  k_gateup<<<g4, 256, 0, stream>>>(x, wg, wu, offsets, counts, list_token, hidden);

  dim3 g5(NEXP * 256, HDIM / 256);
  k_down<<<g5, 256, 0, stream>>>(hidden, wd, offsets, counts, list_token, list_w, out);
}
